// LatticeNode_40192303956690
// MI455X (gfx1250) — compile-verified, matrix operand order checked by reading, not run on hardware
//
#include <hip/hip_runtime.h>
#include <hip/hip_bf16.h>
#include <math.h>

// ---------------------------------------------------------------------------
// CDNA5 (gfx1250) fused LatticeNode update.
// Strategy: bf16 WMMA (v_wmma_f32_16x16x32_bf16) for all GEMMs, f32 accum.
// Weights pre-swizzled into per-lane B-fragment order so each B operand is one
// contiguous 32B load per lane. One wave owns one 16-row tile end-to-end.
// Hot GEMM loops process two 16-col tiles per iteration (two independent
// accumulator chains) to hide WMMA->VALU hazard NOPs and feed the XDL pipe.
// ---------------------------------------------------------------------------

typedef __bf16 bf16_t;
typedef bf16_t v16bf __attribute__((ext_vector_type(16)));
typedef float  v8f   __attribute__((ext_vector_type(8)));

#define F_DIM 128
#define R_DIM 64
#define WAVES_PER_BLOCK 8
#define SMEM_PER_WAVE   (28 * 1024)   // 4KB bf16 tbuf + 3x 8KB f32 buffers

__device__ __forceinline__ int imin(int a, int b) { return a < b ? a : b; }
__device__ __forceinline__ int imax(int a, int b) { return a > b ? a : b; }

__device__ __forceinline__ float ssilu(float v) {
  // silu(x) * (1/0.6); v_rcp_f32 instead of IEEE divide (bf16-operand noise
  // floor is far above 1-ulp rcp error).
  return (1.0f / 0.6f) * v * __builtin_amdgcn_rcpf(1.0f + __expf(-v));
}

__device__ __forceinline__ v8f wmma_acc(v16bf a, v16bf b, v8f c) {
  // D = A(16x32 bf16) * B(32x16 bf16) + C(16x16 f32)
  return __builtin_amdgcn_wmma_f32_16x16x32_bf16(false, a, false, b,
                                                 (short)0, c, false, false);
}

// ---- A-fragment builders (ISA 7.12.2 16-bit A 16x32 layout) ----------------
// lane holds row M = lane&15; K offsets: lanes 0-15 -> {0..7,16..23},
// lanes 16-31 -> {8..15,24..31} (kb = +8).
__device__ __forceinline__ v16bf afrag_f32(const float* rowp, int k0, int hi) {
  int kb = k0 + (hi ? 8 : 0);
  v16bf a;
#pragma unroll
  for (int j = 0; j < 8; ++j) {
    a[j]     = (bf16_t)rowp[kb + j];
    a[j + 8] = (bf16_t)rowp[kb + 16 + j];
  }
  return a;
}
__device__ __forceinline__ v16bf afrag_sum(const float* r0, const float* r1,
                                           int k0, int hi) {
  int kb = k0 + (hi ? 8 : 0);
  v16bf a;
#pragma unroll
  for (int j = 0; j < 8; ++j) {
    a[j]     = (bf16_t)(r0[kb + j] + r1[kb + j]);
    a[j + 8] = (bf16_t)(r0[kb + 16 + j] + r1[kb + 16 + j]);
  }
  return a;
}
__device__ __forceinline__ v16bf afrag_bf16(const bf16_t* rowp, int k0, int hi) {
  int kb = k0 + (hi ? 8 : 0);
  v16bf a;
#pragma unroll
  for (int j = 0; j < 8; ++j) {
    a[j]     = rowp[kb + j];
    a[j + 8] = rowp[kb + 16 + j];
  }
  return a;
}

// ---- B-fragment: pre-swizzled weights, 16 contiguous bf16 per lane ---------
__device__ __forceinline__ v16bf bfrag(const unsigned short* W, int ksteps,
                                       int nt, int ks, int lane) {
  const v16bf* p =
      (const v16bf*)(W + ((size_t)((nt * ksteps + ks) * 32 + lane) << 4));
  return *p;
}

__device__ __forceinline__ v8f dinit(const float* bias, int nt, int col) {
  float bv = bias[nt * 16 + col];
  v8f c;
#pragma unroll
  for (int r = 0; r < 8; ++r) c[r] = bv;
  return c;
}
__device__ __forceinline__ v8f dzero() {
  v8f c;
#pragma unroll
  for (int r = 0; r < 8; ++r) c[r] = 0.0f;
  return c;
}

// ---------------------------------------------------------------------------
// Prep: convert f32 weight [K,N] -> bf16 B-fragment layout.
// flat index t: j=t&15 (frag elem), lane=(t>>4)&31, then ks, nt.
// value (k,n): n = nt*16 + (lane&15); k = ks*32 + (lane>=16?8:0) + j + (j>=8?8:0)
// ---------------------------------------------------------------------------
__global__ __launch_bounds__(256) void prep_weight(const float* __restrict__ W,
                                                   unsigned short* __restrict__ out,
                                                   int K, int Ncols) {
  int ksteps = K / 32;
  size_t total = (size_t)K * Ncols;
  for (size_t t = (size_t)blockIdx.x * blockDim.x + threadIdx.x; t < total;
       t += (size_t)gridDim.x * blockDim.x) {
    int j = (int)(t & 15);
    int lane = (int)((t >> 4) & 31);
    size_t rest = t >> 9;
    int ks = (int)(rest % ksteps);
    int nt = (int)(rest / ksteps);
    int n = nt * 16 + (lane & 15);
    int k = ks * 32 + ((lane >> 4) ? 8 : 0) + j + (j >= 8 ? 8 : 0);
    bf16_t v = (bf16_t)W[(size_t)k * Ncols + n];
    out[t] = *(const unsigned short*)&v;
  }
}

// ---------------------------------------------------------------------------
// Node phase: one wave per 16-node tile; fused GEMM chain.
// ---------------------------------------------------------------------------
__global__ __launch_bounds__(256) void node_kernel(
    const float* __restrict__ x, const float* __restrict__ scalar_l,
    const float* __restrict__ vec, const float* __restrict__ vector_l,
    const float* __restrict__ edge_feat, const float* __restrict__ edge_udiff,
    const int* __restrict__ batch,
    const unsigned short* Wxp1f, const float* bxp1,
    const unsigned short* Wxp2f, const float* bxp2,
    const unsigned short* Wepf,  const float* bep,
    const unsigned short* Wsl1f, const float* bsl1,
    const unsigned short* Wsl2f, const float* bsl2,
    const unsigned short* Wvlf,
    float* __restrict__ out_hx, float* __restrict__ out_hvec, int N) {
  extern __shared__ __align__(16) char smem[];
  const int wave = threadIdx.x >> 5;
  const int lane = threadIdx.x & 31;
  const int nl = lane & 15;      // column within 16-tile / A-row
  const int hi = lane >> 4;      // lane half selects K sub-range / M offset

  char* wsm = smem + (size_t)wave * SMEM_PER_WAVE;
  bf16_t* tbuf = (bf16_t*)wsm;                  // [16][128] bf16 (t1 / t2)
  float* bufA = (float*)(wsm + 4 * 1024);       // [16][128] f32 (x1)
  float* bufB = (float*)(wsm + 12 * 1024);      // [16][128] f32 (x2)
  float* bufC = (float*)(wsm + 20 * 1024);      // [16][128] f32 (xn / vecn)

  const int tile = blockIdx.x * WAVES_PER_BLOCK + wave;
  const int m0 = tile * 16;
  if (m0 >= N) return;  // wave-uniform -> EXEC stays all-ones for WMMA below

  const int row = m0 + nl;
  const int rowc = imin(row, N - 1);
  const int bat = batch[rowc];

  const float inv_sqrt3 = 0.5773502691896258f;
  const float inv_sqrth = 0.08838834764831845f;  // 1/sqrt(128)

  // warm next tile's rows through L2 (global_prefetch_b8)
  __builtin_prefetch(x + (size_t)imin(rowc + 16 * WAVES_PER_BLOCK, N - 1) * F_DIM, 0, 0);
  __builtin_prefetch(vec + (size_t)rowc * 3 * F_DIM, 0, 0);

  // ---- GEMM1: t1 = ssilu(x @ W_xp1 + b_xp1) -> tbuf (bf16) ----
  v16bf ax[4];
#pragma unroll
  for (int ks = 0; ks < 4; ++ks)
    ax[ks] = afrag_f32(x + (size_t)rowc * F_DIM, ks * 32, hi);
  for (int nt = 0; nt < 8; nt += 2) {
    v8f c0 = dinit(bxp1, nt, nl);
    v8f c1 = dinit(bxp1, nt + 1, nl);
#pragma unroll
    for (int ks = 0; ks < 4; ++ks) {
      c0 = wmma_acc(ax[ks], bfrag(Wxp1f, 4, nt, ks, lane), c0);
      c1 = wmma_acc(ax[ks], bfrag(Wxp1f, 4, nt + 1, ks, lane), c1);
    }
#pragma unroll
    for (int r = 0; r < 8; ++r) {
      tbuf[(r + 8 * hi) * F_DIM + nt * 16 + nl]       = (bf16_t)ssilu(c0[r]);
      tbuf[(r + 8 * hi) * F_DIM + (nt + 1) * 16 + nl] = (bf16_t)ssilu(c1[r]);
    }
  }

  // ---- GEMM2: x_p = t1 @ W_xp2 + b ; edge_p = e @ W_ep + b ; combine ----
  v16bf at[4];
#pragma unroll
  for (int ks = 0; ks < 4; ++ks)
    at[ks] = afrag_bf16(tbuf + nl * F_DIM, ks * 32, hi);
  v16bf ae[2];
#pragma unroll
  for (int ks = 0; ks < 2; ++ks)
    ae[ks] = afrag_f32(edge_feat + (size_t)rowc * R_DIM, ks * 32, hi);

  // per-subtile epilogue: route x1/x2/xn chunks into LDS buffers
  auto g2_sink = [&](int nt_, const v8f& cx, const v8f& ce) {
    int colb = (nt_ & 7) * 16 + nl;
#pragma unroll
    for (int r = 0; r < 8; ++r) {
      float p = cx[r] * ce[r] * inv_sqrt3;
      int M = r + 8 * hi;
      if (nt_ < 8) {
        bufA[M * F_DIM + colb] = p;               // x1
      } else if (nt_ < 16) {
        bufB[M * F_DIM + colb] = p;               // x2
      } else {
        int rr = imin(m0 + M, N - 1);
        bufC[M * F_DIM + colb] = p + x[(size_t)rr * F_DIM + colb];  // xn
      }
    }
  };
  for (int nt = 0; nt < 24; nt += 2) {
    v8f cx0 = dinit(bxp2, nt, nl);
    v8f cx1 = dinit(bxp2, nt + 1, nl);
#pragma unroll
    for (int ks = 0; ks < 4; ++ks) {
      cx0 = wmma_acc(at[ks], bfrag(Wxp2f, 4, nt, ks, lane), cx0);
      cx1 = wmma_acc(at[ks], bfrag(Wxp2f, 4, nt + 1, ks, lane), cx1);
    }
    v8f ce0 = dinit(bep, nt, nl);
    v8f ce1 = dinit(bep, nt + 1, nl);
#pragma unroll
    for (int ks = 0; ks < 2; ++ks) {
      ce0 = wmma_acc(ae[ks], bfrag(Wepf, 2, nt, ks, lane), ce0);
      ce1 = wmma_acc(ae[ks], bfrag(Wepf, 2, nt + 1, ks, lane), ce1);
    }
    g2_sink(nt, cx0, ce0);
    g2_sink(nt + 1, cx1, ce1);
  }

  // ---- GEMM3: t2 = ssilu([xn | scalar_l[batch]] @ W_sl1 + b) -> tbuf ----
  v16bf ah[8];
#pragma unroll
  for (int ks = 0; ks < 4; ++ks)
    ah[ks] = afrag_f32(bufC + nl * F_DIM, ks * 32, hi);
  const float* slrow = scalar_l + (size_t)bat * F_DIM;
#pragma unroll
  for (int ks = 0; ks < 4; ++ks)
    ah[4 + ks] = afrag_f32(slrow, ks * 32, hi);
  for (int nt = 0; nt < 8; nt += 2) {
    v8f c0 = dinit(bsl1, nt, nl);
    v8f c1 = dinit(bsl1, nt + 1, nl);
#pragma unroll
    for (int ks = 0; ks < 8; ++ks) {
      c0 = wmma_acc(ah[ks], bfrag(Wsl1f, 8, nt, ks, lane), c0);
      c1 = wmma_acc(ah[ks], bfrag(Wsl1f, 8, nt + 1, ks, lane), c1);
    }
#pragma unroll
    for (int r = 0; r < 8; ++r) {
      tbuf[(r + 8 * hi) * F_DIM + nt * 16 + nl]       = (bf16_t)ssilu(c0[r]);
      tbuf[(r + 8 * hi) * F_DIM + (nt + 1) * 16 + nl] = (bf16_t)ssilu(c1[r]);
    }
  }

  // ---- GEMM4: hx = ssilu(t2 @ W_sl2 + b) + xn ----
#pragma unroll
  for (int ks = 0; ks < 4; ++ks)
    at[ks] = afrag_bf16(tbuf + nl * F_DIM, ks * 32, hi);
  auto g4_sink = [&](int nt_, const v8f& c) {
    int col = nt_ * 16 + nl;
#pragma unroll
    for (int r = 0; r < 8; ++r) {
      int M = r + 8 * hi;
      int rr = m0 + M;
      float hv = ssilu(c[r]) + bufC[M * F_DIM + col];
      if (rr < N) out_hx[(size_t)rr * F_DIM + col] = hv;
    }
  };
  for (int nt = 0; nt < 8; nt += 2) {
    v8f c0 = dinit(bsl2, nt, nl);
    v8f c1 = dinit(bsl2, nt + 1, nl);
#pragma unroll
    for (int ks = 0; ks < 4; ++ks) {
      c0 = wmma_acc(at[ks], bfrag(Wsl2f, 4, nt, ks, lane), c0);
      c1 = wmma_acc(at[ks], bfrag(Wsl2f, 4, nt + 1, ks, lane), c1);
    }
    g4_sink(nt, c0);
    g4_sink(nt + 1, c1);
  }

  // ---- hvec: per axis, vecn = (x1*vec + x2*udiff)/sqrt(F); ----
  // ---- hvec = (vecn + vector_l[batch]) @ W_vl + vecn          ----
  for (int a = 0; a < 3; ++a) {
    for (int idx = lane; idx < 16 * F_DIM; idx += 32) {
      int m = idx >> 7, cc = idx & 127;
      int rr = imin(m0 + m, N - 1);
      float vn = (bufA[idx] * vec[((size_t)rr * 3 + a) * F_DIM + cc] +
                  bufB[idx] * edge_udiff[(size_t)rr * 3 + a]) * inv_sqrth;
      bufC[idx] = vn;  // reuse xn buffer (consumed)
    }
    const float* vlrow = vector_l + ((size_t)bat * 3 + a) * F_DIM;
    v16bf av[4];
#pragma unroll
    for (int ks = 0; ks < 4; ++ks)
      av[ks] = afrag_sum(bufC + nl * F_DIM, vlrow, ks * 32, hi);
    auto hv_sink = [&](int nt_, const v8f& c) {
      int col = nt_ * 16 + nl;
#pragma unroll
      for (int r = 0; r < 8; ++r) {
        int M = r + 8 * hi;
        int rr = m0 + M;
        float hv = c[r] + bufC[M * F_DIM + col];
        if (rr < N) out_hvec[((size_t)rr * 3 + a) * F_DIM + col] = hv;
      }
    };
    for (int nt = 0; nt < 8; nt += 2) {
      v8f c0 = dzero();
      v8f c1 = dzero();
#pragma unroll
      for (int ks = 0; ks < 4; ++ks) {
        c0 = wmma_acc(av[ks], bfrag(Wvlf, 4, nt, ks, lane), c0);
        c1 = wmma_acc(av[ks], bfrag(Wvlf, 4, nt + 1, ks, lane), c1);
      }
      hv_sink(nt, c0);
      hv_sink(nt + 1, c1);
    }
  }
}

// ---------------------------------------------------------------------------
// Segment means over sorted batch: one block per batch id, binary search.
// ---------------------------------------------------------------------------
__device__ __forceinline__ int lower_bound_i(const int* a, int n, int v) {
  int lo = 0, hi = n;
  while (lo < hi) {
    int m = (lo + hi) >> 1;
    if (a[m] < v) lo = m + 1; else hi = m;
  }
  return lo;
}

__global__ __launch_bounds__(256) void seg_mean_kernel(
    const float* __restrict__ hx, const float* __restrict__ hvec,
    const int* __restrict__ batch, float* __restrict__ mean_x,
    float* __restrict__ mean_vec, int N) {
  int b = blockIdx.x;
  int lo = lower_bound_i(batch, N, b);
  int hi = lower_bound_i(batch, N, b + 1);
  float inv = 1.0f / (float)imax(hi - lo, 1);
  for (int c = threadIdx.x; c < F_DIM; c += blockDim.x) {
    float s = 0.0f;
    for (int r = lo; r < hi; ++r) s += hx[(size_t)r * F_DIM + c];
    mean_x[(size_t)b * F_DIM + c] = s * inv;
  }
  for (int idx = threadIdx.x; idx < 3 * F_DIM; idx += blockDim.x) {
    float s = 0.0f;
    for (int r = lo; r < hi; ++r) s += hvec[(size_t)r * 3 * F_DIM + idx];
    mean_vec[(size_t)b * 3 * F_DIM + idx] = s * inv;
  }
}

// ---------------------------------------------------------------------------
// Global phase (B rows): one wave per 16-row tile.
// ---------------------------------------------------------------------------
__global__ __launch_bounds__(256) void global_kernel(
    const float* __restrict__ scalar_l, const float* __restrict__ vector_l,
    const float* __restrict__ mean_x, const float* __restrict__ mean_vec,
    const unsigned short* Wsg1f, const float* bsg1,
    const unsigned short* Wsg2f, const float* bsg2,
    const unsigned short* Wvgf, const unsigned short* Wvpf,
    const unsigned short* Wlp1f, const float* blp1,
    const unsigned short* Wlp2f, const float* blp2,
    float* sl_ws, float* vl_ws, float* h1_ws,
    float* __restrict__ out_sl, float* __restrict__ out_vl, int B) {
  extern __shared__ __align__(16) char smem[];
  const int wave = threadIdx.x >> 5;
  const int lane = threadIdx.x & 31;
  const int nl = lane & 15;
  const int hi = lane >> 4;

  char* wsm = smem + (size_t)wave * SMEM_PER_WAVE;
  bf16_t* tbuf = (bf16_t*)wsm;              // [16][128] bf16
  float* bufA = (float*)(wsm + 4 * 1024);   // vl tile, later s1
  float* bufB = (float*)(wsm + 12 * 1024);  // vnorm accum, later s3
  float* bufC = (float*)(wsm + 20 * 1024);  // s2

  const int tile = blockIdx.x * WAVES_PER_BLOCK + wave;
  const int m0 = tile * 16;
  if (m0 >= B) return;
  const int row = m0 + nl;

  // ---- sl = scalar_l + ssilu(ssilu([mean_x|scalar_l] @ Wsg1 + b) @ Wsg2 + b)
  v16bf ag[8];
#pragma unroll
  for (int ks = 0; ks < 4; ++ks)
    ag[ks] = afrag_f32(mean_x + (size_t)row * F_DIM, ks * 32, hi);
#pragma unroll
  for (int ks = 0; ks < 4; ++ks)
    ag[4 + ks] = afrag_f32(scalar_l + (size_t)row * F_DIM, ks * 32, hi);
  for (int nt = 0; nt < 8; nt += 2) {
    v8f c0 = dinit(bsg1, nt, nl);
    v8f c1 = dinit(bsg1, nt + 1, nl);
#pragma unroll
    for (int ks = 0; ks < 8; ++ks) {
      c0 = wmma_acc(ag[ks], bfrag(Wsg1f, 8, nt, ks, lane), c0);
      c1 = wmma_acc(ag[ks], bfrag(Wsg1f, 8, nt + 1, ks, lane), c1);
    }
#pragma unroll
    for (int r = 0; r < 8; ++r) {
      tbuf[(r + 8 * hi) * F_DIM + nt * 16 + nl]       = (bf16_t)ssilu(c0[r]);
      tbuf[(r + 8 * hi) * F_DIM + (nt + 1) * 16 + nl] = (bf16_t)ssilu(c1[r]);
    }
  }
  v16bf at[4];
#pragma unroll
  for (int ks = 0; ks < 4; ++ks)
    at[ks] = afrag_bf16(tbuf + nl * F_DIM, ks * 32, hi);
  for (int nt = 0; nt < 8; ++nt) {
    v8f c = dinit(bsg2, nt, nl);
#pragma unroll
    for (int ks = 0; ks < 4; ++ks)
      c = wmma_acc(at[ks], bfrag(Wsg2f, 4, nt, ks, lane), c);
    int col = nt * 16 + nl;
#pragma unroll
    for (int r = 0; r < 8; ++r) {
      int M = r + 8 * hi, rr = m0 + M;
      sl_ws[(size_t)rr * F_DIM + col] =
          scalar_l[(size_t)rr * F_DIM + col] + ssilu(c[r]);
    }
  }

  // ---- vl = vector_l + (mean_vec+vector_l)@Wvg ; h1/h2 = vl@Wvp ; vnorm ----
  for (int idx = lane; idx < 16 * F_DIM; idx += 32) bufB[idx] = 0.0f;
  for (int a = 0; a < 3; ++a) {
    const float* mvrow = mean_vec + ((size_t)row * 3 + a) * F_DIM;
    const float* vlrow = vector_l + ((size_t)row * 3 + a) * F_DIM;
    v16bf av[4];
#pragma unroll
    for (int ks = 0; ks < 4; ++ks)
      av[ks] = afrag_sum(mvrow, vlrow, ks * 32, hi);
    for (int nt = 0; nt < 8; ++nt) {
      v8f c = dzero();
#pragma unroll
      for (int ks = 0; ks < 4; ++ks)
        c = wmma_acc(av[ks], bfrag(Wvgf, 4, nt, ks, lane), c);
      int col = nt * 16 + nl;
#pragma unroll
      for (int r = 0; r < 8; ++r) {
        int M = r + 8 * hi, rr = m0 + M;
        float v = c[r] + vector_l[((size_t)rr * 3 + a) * F_DIM + col];
        vl_ws[((size_t)rr * 3 + a) * F_DIM + col] = v;
        bufA[M * F_DIM + col] = v;
      }
    }
    v16bf avl[4];
#pragma unroll
    for (int ks = 0; ks < 4; ++ks)
      avl[ks] = afrag_f32(bufA + nl * F_DIM, ks * 32, hi);
    for (int nt = 0; nt < 16; ++nt) {
      v8f c = dzero();
#pragma unroll
      for (int ks = 0; ks < 4; ++ks)
        c = wmma_acc(avl[ks], bfrag(Wvpf, 4, nt, ks, lane), c);
      int col = (nt & 7) * 16 + nl;
#pragma unroll
      for (int r = 0; r < 8; ++r) {
        int M = r + 8 * hi, rr = m0 + M;
        if (nt < 8)
          h1_ws[((size_t)rr * 3 + a) * F_DIM + col] = c[r];
        else
          bufB[M * F_DIM + col] += c[r] * c[r];
      }
    }
  }
  for (int idx = lane; idx < 16 * F_DIM; idx += 32)
    bufB[idx] = sqrtf(bufB[idx] + 1e-8f);

  // ---- sh = ssilu([sl|vnorm] @ Wlp1 + b) @ Wlp2 + b -> s1,s2,s3 ----
  v16bf as[8];
#pragma unroll
  for (int ks = 0; ks < 4; ++ks)
    as[ks] = afrag_f32(sl_ws + (size_t)row * F_DIM, ks * 32, hi);
#pragma unroll
  for (int ks = 0; ks < 4; ++ks)
    as[4 + ks] = afrag_f32(bufB + nl * F_DIM, ks * 32, hi);
  for (int nt = 0; nt < 8; ++nt) {
    v8f c = dinit(blp1, nt, nl);
#pragma unroll
    for (int ks = 0; ks < 8; ++ks)
      c = wmma_acc(as[ks], bfrag(Wlp1f, 8, nt, ks, lane), c);
#pragma unroll
    for (int r = 0; r < 8; ++r)
      tbuf[(r + 8 * hi) * F_DIM + nt * 16 + nl] = (bf16_t)ssilu(c[r]);
  }
#pragma unroll
  for (int ks = 0; ks < 4; ++ks)
    at[ks] = afrag_bf16(tbuf + nl * F_DIM, ks * 32, hi);
  for (int nt = 0; nt < 24; ++nt) {
    v8f c = dinit(blp2, nt, nl);
#pragma unroll
    for (int ks = 0; ks < 4; ++ks)
      c = wmma_acc(at[ks], bfrag(Wlp2f, 4, nt, ks, lane), c);
    int col = (nt % 8) * 16 + nl;
#pragma unroll
    for (int r = 0; r < 8; ++r) {
      int M = r + 8 * hi;
      if (nt < 8)       bufA[M * F_DIM + col] = c[r];  // s1
      else if (nt < 16) bufC[M * F_DIM + col] = c[r];  // s2
      else              bufB[M * F_DIM + col] = c[r];  // s3
    }
  }

  // ---- sl_out = s2 + sl*tanh(s3) ; vl_out = s1*h1 + vl ----
  for (int idx = lane; idx < 16 * F_DIM; idx += 32) {
    int m = idx >> 7, cc = idx & 127;
    int rr = m0 + m;
    float slv = sl_ws[(size_t)rr * F_DIM + cc];
    out_sl[(size_t)rr * F_DIM + cc] = bufC[idx] + slv * tanhf(bufB[idx]);
  }
  for (int a = 0; a < 3; ++a) {
    for (int idx = lane; idx < 16 * F_DIM; idx += 32) {
      int m = idx >> 7, cc = idx & 127;
      int rr = m0 + m;
      size_t o = ((size_t)rr * 3 + a) * F_DIM + cc;
      out_vl[o] = bufA[(m * F_DIM) + cc] * h1_ws[o] + vl_ws[o];
    }
  }
}

// ---------------------------------------------------------------------------
// l_delta = vl_out @ W_ml  -> [B,3,1]
// ---------------------------------------------------------------------------
__global__ __launch_bounds__(256) void ldelta_kernel(
    const float* __restrict__ vl_out, const float* __restrict__ W_ml,
    float* __restrict__ out_ld, int total) {
  int t = blockIdx.x * blockDim.x + threadIdx.x;
  if (t >= total) return;
  float s = 0.0f;
  for (int c = 0; c < F_DIM; ++c) s += vl_out[(size_t)t * F_DIM + c] * W_ml[c];
  out_ld[t] = s;
}

// ---------------------------------------------------------------------------
extern "C" void kernel_launch(void* const* d_in, const int* in_sizes, int n_in,
                              void* d_out, int out_size, void* d_ws,
                              size_t ws_size, hipStream_t stream) {
  const int F = F_DIM;
  const int N = in_sizes[0] / F;
  const int B = in_sizes[1] / F;

  const float* x         = (const float*)d_in[0];
  const float* scalar_l  = (const float*)d_in[1];
  const float* vec       = (const float*)d_in[2];
  const float* vector_l  = (const float*)d_in[3];
  const float* edge_feat = (const float*)d_in[4];
  const float* edge_ud   = (const float*)d_in[5];
  const int*   batch     = (const int*)d_in[6];
  const float* W_sg1 = (const float*)d_in[7];   const float* b_sg1 = (const float*)d_in[8];
  const float* W_sg2 = (const float*)d_in[9];   const float* b_sg2 = (const float*)d_in[10];
  const float* W_sl1 = (const float*)d_in[11];  const float* b_sl1 = (const float*)d_in[12];
  const float* W_sl2 = (const float*)d_in[13];  const float* b_sl2 = (const float*)d_in[14];
  const float* W_vg  = (const float*)d_in[15];
  const float* W_vl  = (const float*)d_in[16];
  const float* W_xp1 = (const float*)d_in[17];  const float* b_xp1 = (const float*)d_in[18];
  const float* W_xp2 = (const float*)d_in[19];  const float* b_xp2 = (const float*)d_in[20];
  const float* W_ep  = (const float*)d_in[21];  const float* b_ep  = (const float*)d_in[22];
  const float* W_vp  = (const float*)d_in[23];
  const float* W_lp1 = (const float*)d_in[24];  const float* b_lp1 = (const float*)d_in[25];
  const float* W_lp2 = (const float*)d_in[26];  const float* b_lp2 = (const float*)d_in[27];
  const float* W_ml  = (const float*)d_in[28];

  // workspace carve-out (256B aligned)
  char* wp = (char*)d_ws;
  auto carve = [&](size_t bytes) -> char* {
    char* p = wp;
    wp += (bytes + 255) & ~(size_t)255;
    return p;
  };
  unsigned short* Wxp1f = (unsigned short*)carve((size_t)128 * 128 * 2);
  unsigned short* Wxp2f = (unsigned short*)carve((size_t)128 * 384 * 2);
  unsigned short* Wepf  = (unsigned short*)carve((size_t)64  * 384 * 2);
  unsigned short* Wsl1f = (unsigned short*)carve((size_t)256 * 128 * 2);
  unsigned short* Wsl2f = (unsigned short*)carve((size_t)128 * 128 * 2);
  unsigned short* Wvlf  = (unsigned short*)carve((size_t)128 * 128 * 2);
  unsigned short* Wsg1f = (unsigned short*)carve((size_t)256 * 128 * 2);
  unsigned short* Wsg2f = (unsigned short*)carve((size_t)128 * 128 * 2);
  unsigned short* Wvgf  = (unsigned short*)carve((size_t)128 * 128 * 2);
  unsigned short* Wvpf  = (unsigned short*)carve((size_t)128 * 256 * 2);
  unsigned short* Wlp1f = (unsigned short*)carve((size_t)256 * 128 * 2);
  unsigned short* Wlp2f = (unsigned short*)carve((size_t)128 * 384 * 2);
  float* mean_x   = (float*)carve((size_t)B * F * 4);
  float* mean_vec = (float*)carve((size_t)B * 3 * F * 4);
  float* sl_ws    = (float*)carve((size_t)B * F * 4);
  float* vl_ws    = (float*)carve((size_t)B * 3 * F * 4);
  float* h1_ws    = (float*)carve((size_t)B * 3 * F * 4);

  float* out_hx   = (float*)d_out;
  float* out_hvec = out_hx + (size_t)N * F;
  float* out_sl   = out_hvec + (size_t)N * 3 * F;
  float* out_vl   = out_sl + (size_t)B * F;
  float* out_ld   = out_vl + (size_t)B * 3 * F;

  auto prep = [&](const float* W, unsigned short* out, int K, int Ncols) {
    size_t total = (size_t)K * Ncols;
    int blocks = (int)((total + 255) / 256);
    prep_weight<<<blocks, 256, 0, stream>>>(W, out, K, Ncols);
  };
  prep(W_xp1, Wxp1f, 128, 128);
  prep(W_xp2, Wxp2f, 128, 384);
  prep(W_ep,  Wepf,  64,  384);
  prep(W_sl1, Wsl1f, 256, 128);
  prep(W_sl2, Wsl2f, 128, 128);
  prep(W_vl,  Wvlf,  128, 128);
  prep(W_sg1, Wsg1f, 256, 128);
  prep(W_sg2, Wsg2f, 128, 128);
  prep(W_vg,  Wvgf,  128, 128);
  prep(W_vp,  Wvpf,  128, 256);
  prep(W_lp1, Wlp1f, 256, 128);
  prep(W_lp2, Wlp2f, 128, 384);

  const size_t smem_bytes = (size_t)WAVES_PER_BLOCK * SMEM_PER_WAVE;  // 224KB

  int node_tiles = (N + 15) / 16;
  int node_blocks = (node_tiles + WAVES_PER_BLOCK - 1) / WAVES_PER_BLOCK;
  node_kernel<<<node_blocks, 256, smem_bytes, stream>>>(
      x, scalar_l, vec, vector_l, edge_feat, edge_ud, batch,
      Wxp1f, b_xp1, Wxp2f, b_xp2, Wepf, b_ep,
      Wsl1f, b_sl1, Wsl2f, b_sl2, Wvlf, out_hx, out_hvec, N);

  seg_mean_kernel<<<B, 256, 0, stream>>>(out_hx, out_hvec, batch, mean_x,
                                         mean_vec, N);

  int g_tiles = (B + 15) / 16;
  int g_blocks = (g_tiles + WAVES_PER_BLOCK - 1) / WAVES_PER_BLOCK;
  global_kernel<<<g_blocks, 256, smem_bytes, stream>>>(
      scalar_l, vector_l, mean_x, mean_vec, Wsg1f, b_sg1, Wsg2f, b_sg2, Wvgf,
      Wvpf, Wlp1f, b_lp1, Wlp2f, b_lp2, sl_ws, vl_ws, h1_ws, out_sl, out_vl, B);

  int ld_total = B * 3;
  ldelta_kernel<<<(ld_total + 255) / 256, 256, 0, stream>>>(out_vl, W_ml,
                                                            out_ld, ld_total);
}
